// ResidualVectorQuantizer_63118839382729
// MI455X (gfx1250) — matrix operand order repaired and checked by reading, not compiled, not run on hardware
//
#include <hip/hip_runtime.h>
#include <hip/hip_bf16.h>

typedef __attribute__((ext_vector_type(16))) _Float16 v16h;
typedef __attribute__((ext_vector_type(8)))  float    v8f;

#define RVQ_B 65536
#define RVQ_L 4
#define RVQ_K 1024
#define RVQ_D 512

#define ROWS_PER_BLOCK 128           // 8 waves * 16 rows
#define RES_STRIDE 516               // 512 + 4 pad floats -> bank spread
#define SMEM_BYTES (ROWS_PER_BLOCK * RES_STRIDE * 4 + ROWS_PER_BLOCK * 4)

// ---------------------------------------------------------------------------
// Kernel 1: pack f32 codebooks [L,K,D] into f16 WMMA B-fragment layout.
// Half index dst = (((l*64 + tile)*16 + chunk)*32 + lane)*16 + j
//   where: n = tile*16 + (lane&15), hi = lane>>4, d = chunk*32 + hi*16 + j
// So a lane's v16h B fragment for (tile,chunk) is one contiguous 32B load.
// ---------------------------------------------------------------------------
__global__ void rvq_pack_codebook(const float* __restrict__ cb,
                                  _Float16* __restrict__ bpack) {
    unsigned id = blockIdx.x * blockDim.x + threadIdx.x;   // 0 .. 2^21-1
    unsigned j     =  id        & 15;
    unsigned lane  = (id >> 4)  & 31;
    unsigned chunk = (id >> 9)  & 15;
    unsigned tile  = (id >> 13) & 63;
    unsigned l     =  id >> 19;
    unsigned n  = tile * 16 + (lane & 15);
    unsigned hi = lane >> 4;
    unsigned d  = chunk * 32 + hi * 16 + j;
    size_t src = ((size_t)l * RVQ_K + n) * RVQ_D + d;
    bpack[id] = (_Float16)cb[src];
}

// ---------------------------------------------------------------------------
// Kernel 2: per-code squared norms (f32) + zero the loss accumulator.
// One wave per codebook row.
// ---------------------------------------------------------------------------
__global__ void rvq_code_norms(const float* __restrict__ cb,
                               float* __restrict__ cnorm,
                               float* __restrict__ lossAcc) {
    if (blockIdx.x == 0 && threadIdx.x == 0) *lossAcc = 0.0f;
    int lane = threadIdx.x & 31;
    int wave = threadIdx.x >> 5;
    int row  = blockIdx.x * 8 + wave;          // 0 .. L*K-1
    if (row >= RVQ_L * RVQ_K) return;
    const float* p = cb + (size_t)row * RVQ_D;
    float s = 0.0f;
    for (int d = lane; d < RVQ_D; d += 32) { float v = p[d]; s += v * v; }
    #pragma unroll
    for (int off = 16; off > 0; off >>= 1) s += __shfl_xor(s, off, 32);
    if (lane == 0) cnorm[row] = s;
}

// ---------------------------------------------------------------------------
// Kernel 3: main RVQ. 256 threads (8 wave32) per block, 128 rows per block.
// Residual kept in f32 in LDS across the 4 levels; distances via
// v_wmma_f32_16x16x32_f16 against the packed f16 codebook.
// ---------------------------------------------------------------------------
__global__ void __launch_bounds__(256)
rvq_main(const float* __restrict__ zE,        // [B,D]
         const float* __restrict__ cbF32,     // [L,K,D]
         const _Float16* __restrict__ bpackH, // packed f16 codebook
         const float* __restrict__ cnorm,     // [L,K]
         float* __restrict__ zqOut,           // [B,D]
         float* __restrict__ codesOut,        // [L,B] (as float)
         float* __restrict__ lossAcc) {
    extern __shared__ float smem[];
    float* res    = smem;                                   // [128][516]
    int*   rowIdx = (int*)(smem + ROWS_PER_BLOCK * RES_STRIDE);

    const int tid  = threadIdx.x;
    const int lane = tid & 31;
    const int wave = tid >> 5;
    const int m    = lane & 15;
    const int hi   = lane >> 4;
    const size_t blockRow0 = (size_t)blockIdx.x * ROWS_PER_BLOCK;

    // ---- load z_e tile into LDS residual (float4, coalesced) ----
    const float4* zin = (const float4*)(zE + blockRow0 * RVQ_D);
    for (int i = tid; i < ROWS_PER_BLOCK * (RVQ_D / 4); i += 256) {
        int row = i >> 7;           // /128 float4 per row
        int c4  = i & 127;
        *(float4*)&res[row * RES_STRIDE + c4 * 4] = zin[i];
    }
    __syncthreads();

    const v16h* bpack = (const v16h*)bpackH;   // v16h index = half index / 16
    float ssq = 0.0f;

    for (int l = 0; l < RVQ_L; ++l) {
        // ---- build A fragments: this wave's 16 rows, all 16 D-chunks ----
        // A 16x32 fragment, k order used here (same as B pack): lane m=lane&15, K(j) = hi*16 + j
        const float* rrow = &res[(wave * 16 + m) * RES_STRIDE];
        v16h a[16];
        #pragma unroll
        for (int c = 0; c < 16; ++c) {
            #pragma unroll
            for (int v = 0; v < 8; ++v) {
                int k0 = hi * 16 + 2 * v;
                float2 f = *(const float2*)&rrow[c * 32 + k0];
                a[c][2 * v]     = (_Float16)f.x;
                a[c][2 * v + 1] = (_Float16)f.y;
            }
        }

        // ---- sweep 64 code tiles of 16: lane-distributed running argmin ----
        float best[8];
        int   bidx[8];
        #pragma unroll
        for (int r = 0; r < 8; ++r) { best[r] = 3.4e38f; bidx[r] = 0; }

        const v16h*  bl = bpack + (size_t)l * 64 * 16 * 32;
        const float* cn_base = cnorm + l * RVQ_K;

        for (int t = 0; t < 64; ++t) {
            float cn = cn_base[t * 16 + m];          // ||c_n||^2, n = t*16+m
            const v16h* bt = bl + (size_t)t * 16 * 32 + lane;
            v8f acc = {};
            #pragma unroll
            for (int c = 0; c < 16; ++c) {
                v16h b = bt[c * 32];
                acc = __builtin_amdgcn_wmma_f32_16x16x32_f16(
                        false, a[c], false, b, (short)0, acc, false, false);
            }
            #pragma unroll
            for (int r = 0; r < 8; ++r) {
                float s = __builtin_fmaf(-2.0f, acc[r], cn);
                if (s < best[r]) { best[r] = s; bidx[r] = t * 16 + m; }
            }
        }

        // ---- cross-lane argmin within each 16-lane half ----
        #pragma unroll
        for (int r = 0; r < 8; ++r) {
            float v = best[r];
            int   ix = bidx[r];
            #pragma unroll
            for (int off = 1; off < 16; off <<= 1) {
                float ov = __shfl_xor(v, off, 32);
                int   oi = __shfl_xor(ix, off, 32);
                if (ov < v || (ov == v && oi < ix)) { v = ov; ix = oi; }
            }
            if (m == 0) {                                  // lanes 0 and 16
                int row = wave * 16 + r + 8 * hi;
                rowIdx[row] = ix;
                codesOut[(size_t)l * RVQ_B + blockRow0 + row] = (float)ix;
            }
        }
        __syncthreads();

        // ---- residual update + loss: gather winning f32 code rows ----
        for (int mm = 0; mm < 16; ++mm) {
            int row = wave * 16 + mm;
            int ix  = rowIdx[row];
            const float* q  = cbF32 + ((size_t)l * RVQ_K + ix) * RVQ_D;
            float*       rp = &res[row * RES_STRIDE];
            for (int d = lane; d < RVQ_D; d += 32) {
                float nr = rp[d] - q[d];
                rp[d] = nr;
                ssq += nr * nr;     // (q - r_prev)^2 == r_new^2, counted x2 later
            }
        }
        __syncthreads();
    }

    // ---- z_q_st = z_e - final_residual (== qsum) ----
    float4* zout = (float4*)(zqOut + blockRow0 * RVQ_D);
    for (int i = tid; i < ROWS_PER_BLOCK * (RVQ_D / 4); i += 256) {
        int row = i >> 7;
        int c4  = i & 127;
        float4 z = zin[i];
        float4 r = *(float4*)&res[row * RES_STRIDE + c4 * 4];
        float4 o; o.x = z.x - r.x; o.y = z.y - r.y; o.z = z.z - r.z; o.w = z.w - r.w;
        zout[i] = o;
    }
    atomicAdd(lossAcc, ssq);
}

// ---------------------------------------------------------------------------
// Kernel 4: finalize loss = 2 * sum(r^2 over all levels) / (B*D)
// ---------------------------------------------------------------------------
__global__ void rvq_finalize(const float* __restrict__ lossAcc,
                             float* __restrict__ out) {
    out[(size_t)RVQ_B * RVQ_D + (size_t)RVQ_L * RVQ_B] =
        2.0f * (*lossAcc) / (float)((size_t)RVQ_B * RVQ_D);
}

extern "C" void kernel_launch(void* const* d_in, const int* in_sizes, int n_in,
                              void* d_out, int out_size, void* d_ws, size_t ws_size,
                              hipStream_t stream) {
    const float* zE    = (const float*)d_in[0];   // [B,D]
    const float* cbF32 = (const float*)d_in[1];   // [L,K,D]

    float* out   = (float*)d_out;
    float* zqOut = out;                                   // B*D
    float* codes = out + (size_t)RVQ_B * RVQ_D;           // L*B
    // loss scalar written by finalize at out[B*D + L*B]

    // workspace layout (d_ws assumed >= ~4.3 MB)
    float*    ws      = (float*)d_ws;
    float*    lossAcc = ws;                               // 1 float (use 64-slot pad)
    float*    cnorm   = ws + 64;                          // L*K floats = 16 KB
    _Float16* bpack   = (_Float16*)(ws + 64 + RVQ_L * RVQ_K); // 4 MB, 32B aligned

    // 1) pack f16 codebook into WMMA B layout
    {
        int total = RVQ_L * RVQ_K * RVQ_D;                // 2^21
        rvq_pack_codebook<<<total / 256, 256, 0, stream>>>(cbF32, bpack);
    }
    // 2) code norms + zero loss accumulator
    {
        int rows = RVQ_L * RVQ_K;                          // 4096, 8 rows/block
        rvq_code_norms<<<rows / 8, 256, 0, stream>>>(cbF32, cnorm, lossAcc);
    }
    // 3) main RVQ
    {
        int blocks = RVQ_B / ROWS_PER_BLOCK;               // 512
        rvq_main<<<blocks, 256, SMEM_BYTES, stream>>>(
            zE, cbF32, bpack, cnorm, zqOut, codes, lossAcc);
    }
    // 4) finalize loss
    rvq_finalize<<<1, 1, 0, stream>>>(lossAcc, out);
}
